// GCN_72164040507402
// MI455X (gfx1250) — matrix operand order repaired and checked by reading, not compiled, not run on hardware
//
#include <hip/hip_runtime.h>

typedef __attribute__((ext_vector_type(16))) _Float16 v16h;
typedef __attribute__((ext_vector_type(8)))  float    v8f;

#define IN_CH  128
#define HID_CH 64
#define OUT_CH 32

// ---------------- utility kernels ----------------

__global__ void k_zero_u32(unsigned* __restrict__ p, int n) {
  int i = blockIdx.x * blockDim.x + threadIdx.x;
  if (i < n) p[i] = 0u;
}

__global__ void k_degree(const int* __restrict__ dst, unsigned* __restrict__ cnt, int E) {
  int i = blockIdx.x * blockDim.x + threadIdx.x;
  if (i < E) atomicAdd(&cnt[dst[i]], 1u);
}

// deg includes the self-loop (+1); every node has deg >= 1 so dis = rsqrt(deg)
__global__ void k_dis(const unsigned* __restrict__ cnt, float* __restrict__ dis, int n) {
  int i = blockIdx.x * blockDim.x + threadIdx.x;
  if (i < n) dis[i] = rsqrtf((float)(cnt[i] + 1u));
}

// Transpose + convert weights to f16: W[k][n] (f32) -> WhT[n][k] (f16)
__global__ void k_cvt_weights(const float* __restrict__ W1, const float* __restrict__ W2,
                              _Float16* __restrict__ W1hT, _Float16* __restrict__ W2hT) {
  int i = blockIdx.x * blockDim.x + threadIdx.x;
  if (i < IN_CH * HID_CH) {
    int k = i / HID_CH, n = i % HID_CH;
    W1hT[n * IN_CH + k] = (_Float16)W1[i];
  } else {
    int j = i - IN_CH * HID_CH;
    if (j < HID_CH * OUT_CH) {
      int k = j / OUT_CH, n = j % OUT_CH;
      W2hT[n * HID_CH + k] = (_Float16)W2[j];
    }
  }
}

// ---------------- WMMA GEMM: C[N x NC] = A[N x K] (@opt ReLU) * B[K x NC] ----------------
// One wave computes a 16-row M tile; NT = NC/16 N-tiles, KC = K/32 K-chunks.
template <int K, int NC, bool RELU>
__global__ __launch_bounds__(256) void k_gemm_wmma(const float* __restrict__ A,
                                                   const _Float16* __restrict__ BT, // [NC][K]
                                                   float* __restrict__ C, int Nrows) {
  const int lane = threadIdx.x & 31;
  const int wave = blockIdx.x * (blockDim.x >> 5) + (threadIdx.x >> 5);
  const int mtiles = (Nrows + 15) >> 4;
  if (wave >= mtiles) return;   // wave-uniform exit: EXEC all-1s for active waves

  const int l15 = lane & 15;
  const int hs  = lane >> 4;    // 0: lanes 0-15, 1: lanes 16-31
  const int r0  = wave << 4;
  int row = r0 + l15;
  if (row >= Nrows) row = Nrows - 1;  // clamp loads (stores guarded)

  constexpr int KC = K / 32;
  constexpr int NT = NC / 16;
  const v8f vzero = {0.f, 0.f, 0.f, 0.f, 0.f, 0.f, 0.f, 0.f};
  v8f acc[NT];
#pragma unroll
  for (int nt = 0; nt < NT; ++nt) acc[nt] = vzero;

#pragma unroll
  for (int kc = 0; kc < KC; ++kc) {
    // ---- A fragment: lane holds K = kb..kb+15 contiguous (kb = 0 / 16), matching B
    const int kbA = kc * 32 + hs * 16;
    float fa[16];
    const float4* p0 = reinterpret_cast<const float4*>(A + (long)row * K + kbA);
    *reinterpret_cast<float4*>(fa + 0) = p0[0];
    *reinterpret_cast<float4*>(fa + 4) = p0[1];
    const float4* p1 = reinterpret_cast<const float4*>(A + (long)row * K + kbA + 8);
    *reinterpret_cast<float4*>(fa + 8)  = p1[0];
    *reinterpret_cast<float4*>(fa + 12) = p1[1];
    v16h a;
#pragma unroll
    for (int i = 0; i < 16; ++i) {
      float v = fa[i];
      if (RELU) v = v > 0.f ? v : 0.f;
      a[i] = (_Float16)v;
    }
    // ---- B fragments: lanes 0-15 hold K=kb2..kb2+15 contiguous (kb2 = 0 / 16)
    const int kbB = kc * 32 + hs * 16;
#pragma unroll
    for (int nt = 0; nt < NT; ++nt) {
      const int col = nt * 16 + l15;
      v16h b = *reinterpret_cast<const v16h*>(BT + (long)col * K + kbB);
      acc[nt] = __builtin_amdgcn_wmma_f32_16x16x32_f16(false, a, false, b,
                                                       (short)0, acc[nt], false, false);
    }
  }

  // ---- store: C/D layout, VGPR r -> M = r0 + hs*8 + r, N = l15 (+16*nt)
  const int mb = r0 + hs * 8;
#pragma unroll
  for (int nt = 0; nt < NT; ++nt) {
    const int col = nt * 16 + l15;
#pragma unroll
    for (int r = 0; r < 8; ++r) {
      int m = mb + r;
      if (m < Nrows) C[(long)m * NC + col] = acc[nt][r];
    }
  }
}

// ---------------- aggregation kernels ----------------

// agg[i][c] = bias[c] + h[i][c] * dis[i]^2   (bias + self-loop term folded in)
template <int C>
__global__ void k_agg_init(const float* __restrict__ h, const float* __restrict__ dis,
                           const float* __restrict__ bias, float* __restrict__ agg, int Nn) {
  long i = (long)blockIdx.x * blockDim.x + threadIdx.x;
  if (i < (long)Nn * C) {
    int node = (int)(i >> (C == 64 ? 6 : 5));
    int c = (int)(i & (C - 1));
    float d = dis[node];
    agg[i] = bias[c] + h[i] * d * d;
  }
}

// agg[dst][c] += h[src][c] * dis[src] * dis[dst]
template <int C>
__global__ void k_agg_edges(const int* __restrict__ src, const int* __restrict__ dst,
                            const float* __restrict__ h, const float* __restrict__ dis,
                            float* __restrict__ agg, int E) {
  long t = (long)blockIdx.x * blockDim.x + threadIdx.x;
  long e = t >> (C == 64 ? 6 : 5);
  int c = (int)(t & (C - 1));
  if (e < (long)E) {
    int s = src[e], d = dst[e];
    float w = dis[s] * dis[d];
    atomicAdd(&agg[(long)d * C + c], h[(long)s * C + c] * w);
  }
}

// ---------------- launcher ----------------

extern "C" void kernel_launch(void* const* d_in, const int* in_sizes, int n_in,
                              void* d_out, int out_size, void* d_ws, size_t ws_size,
                              hipStream_t stream) {
  const float* x  = (const float*)d_in[0];
  const int*   ei = (const int*)d_in[1];
  const float* W1 = (const float*)d_in[2];
  const float* b1 = (const float*)d_in[3];
  const float* W2 = (const float*)d_in[4];
  const float* b2 = (const float*)d_in[5];
  float* out = (float*)d_out;

  const int Nn = in_sizes[0] / IN_CH;   // 100000
  const int E  = in_sizes[1] / 2;       // 1600000
  const int* srcA = ei;                 // edge_index[0]
  const int* dstA = ei + E;             // edge_index[1]

  // workspace carve-out (256B aligned slices)
  char* w = (char*)d_ws;
  size_t off = 0;
  auto alloc = [&](size_t bytes) -> void* {
    void* p = (void*)(w + off);
    off += (bytes + 255) & ~(size_t)255;
    return p;
  };
  unsigned*  cnt  = (unsigned*)alloc((size_t)Nn * 4);
  float*     dis  = (float*)alloc((size_t)Nn * 4);
  float*     h1   = (float*)alloc((size_t)Nn * HID_CH * 4);
  float*     agg1 = (float*)alloc((size_t)Nn * HID_CH * 4);
  float*     h2   = (float*)alloc((size_t)Nn * OUT_CH * 4);
  _Float16*  W1hT = (_Float16*)alloc((size_t)IN_CH * HID_CH * 2);
  _Float16*  W2hT = (_Float16*)alloc((size_t)HID_CH * OUT_CH * 2);
  (void)ws_size; (void)n_in; (void)out_size;

  const int B = 256;

  // degree -> dis
  k_zero_u32<<<(Nn + B - 1) / B, B, 0, stream>>>(cnt, Nn);
  k_degree<<<(E + B - 1) / B, B, 0, stream>>>(dstA, cnt, E);
  k_dis<<<(Nn + B - 1) / B, B, 0, stream>>>(cnt, dis, Nn);

  // weights -> f16 transposed
  k_cvt_weights<<<(IN_CH * HID_CH + HID_CH * OUT_CH + B - 1) / B, B, 0, stream>>>(W1, W2, W1hT, W2hT);

  const int mtiles = (Nn + 15) / 16;
  const int ggemm  = (mtiles + 7) / 8;   // 8 waves / 256-thread block

  // layer 1: h1 = x @ W1 ; agg1 = b1 + self + scatter(edges)
  k_gemm_wmma<IN_CH, HID_CH, false><<<ggemm, 256, 0, stream>>>(x, W1hT, h1, Nn);
  long t1 = (long)Nn * HID_CH;
  k_agg_init<HID_CH><<<(int)((t1 + B - 1) / B), B, 0, stream>>>(h1, dis, b1, agg1, Nn);
  long te1 = (long)E * HID_CH;
  k_agg_edges<HID_CH><<<(int)((te1 + B - 1) / B), B, 0, stream>>>(srcA, dstA, h1, dis, agg1, E);

  // layer 2: h2 = relu(agg1) @ W2 (ReLU fused into A-load) ; out = b2 + self + scatter
  k_gemm_wmma<HID_CH, OUT_CH, true><<<ggemm, 256, 0, stream>>>(agg1, W2hT, h2, Nn);
  long t2 = (long)Nn * OUT_CH;
  k_agg_init<OUT_CH><<<(int)((t2 + B - 1) / B), B, 0, stream>>>(h2, dis, b2, out, Nn);
  long te2 = (long)E * OUT_CH;
  k_agg_edges<OUT_CH><<<(int)((te2 + B - 1) / B), B, 0, stream>>>(srcA, dstA, h2, dis, out, E);
}
